// BlocksGNN_34505767256697
// MI455X (gfx1250) — matrix operand order repaired and checked by reading, not compiled, not run on hardware
//
#include <hip/hip_runtime.h>
#include <hip/hip_bf16.h>

typedef __attribute__((ext_vector_type(16))) __bf16 v16bf;
typedef __attribute__((ext_vector_type(8)))  __bf16 v8bf;
typedef __attribute__((ext_vector_type(8)))  float  v8f;

union AFrag { v16bf v; v8bf h[2]; };

__device__ __forceinline__ AFrag load_a_frag(const __bf16* ap, int koff) {
    AFrag f;
    f.h[0] = *(const v8bf*)(ap + koff);
    f.h[1] = *(const v8bf*)(ap + 8 + koff);
    return f;
}

__device__ __forceinline__ void load_b_frags(const __bf16* __restrict__ W, int kb,
                                             int wave, int lane, v16bf* b) {
#pragma unroll
    for (int t = 0; t < 4; ++t)
        b[t] = *(const v16bf*)(W + ((long)kb * 32 + wave * 4 + t) * 512 + lane * 16);
}

__device__ __forceinline__ v8f wmma_bf16(const AFrag& a, v16bf b, v8f c) {
    return __builtin_amdgcn_wmma_f32_16x16x32_bf16(false, a.v, false, b, (short)0, c, false, false);
}

// ---------------------------------------------------------------------------
// f32 -> bf16 elementwise convert
// ---------------------------------------------------------------------------
__global__ __launch_bounds__(256) void cvt_bf16_kernel(const float* __restrict__ x,
                                                       __bf16* __restrict__ y, long nElem) {
    long idx = (long)blockIdx.x * 256 + threadIdx.x;
    if (idx < nElem) y[idx] = (__bf16)x[idx];
}

// ---------------------------------------------------------------------------
// Repack a row-major [K,N] f32 weight into WMMA B-fragment-major bf16:
// fragment (kb,nb) covers K in [32kb,32kb+32), N in [16nb,16nb+16).
// Within a fragment: lane l holds 16 contiguous bf16 = B[32kb+16*(l>>4)+i][16nb+(l&15)].
// ---------------------------------------------------------------------------
__global__ __launch_bounds__(256) void pack_w_kernel(const float* __restrict__ W,
                                                     __bf16* __restrict__ P, int K, int N) {
    long idx = (long)blockIdx.x * 256 + threadIdx.x;
    long total = (long)K * N;
    if (idx >= total) return;
    long frag   = idx >> 9;
    int  within = (int)(idx & 511);
    int  lane   = within >> 4;
    int  i      = within & 15;
    int  nbCount = N >> 4;
    int  kb = (int)(frag / nbCount);
    int  nb = (int)(frag % nbCount);
    int  nc = nb * 16 + (lane & 15);
    int  kc = kb * 32 + (lane >> 4) * 16 + i;
    P[idx] = (__bf16)W[(long)kc * N + nc];
}

// ---------------------------------------------------------------------------
// agg[v][c] = sum_{j=0..6} edge_attr[v*7+j][c]   (edges row-major (i,j))
// ---------------------------------------------------------------------------
__global__ __launch_bounds__(256) void agg_kernel(const __bf16* __restrict__ edge,
                                                  __bf16* __restrict__ agg, long nElem) {
    long idx = (long)blockIdx.x * 256 + threadIdx.x;
    if (idx >= nElem) return;
    long v = idx >> 9;
    int  c = (int)(idx & 511);
    const __bf16* p = edge + v * 7 * 512 + c;
    float s = 0.f;
#pragma unroll
    for (int j = 0; j < 7; ++j) s += (float)p[j * 512];
    agg[idx] = (__bf16)s;
}

// ---------------------------------------------------------------------------
// Fused MLP3:  Linear(1024->512)+ReLU -> Linear(512->512) -> LayerNorm -> ReLU
//              -> Linear(512->512), bf16 in / bf16 out, f32 accumulate.
// One workgroup = 16 rows x 512 cols.  8 waves; wave w owns cols [64w,64w+64).
// K-loops are software-pipelined (double-buffered A/B fragments) so
// global_load_b128 of iteration k+1/k+2 overlaps the v_wmma of iteration k.
// ---------------------------------------------------------------------------
__global__ __launch_bounds__(256) void fused_mlp3(
    const __bf16* __restrict__ src0, const __bf16* __restrict__ src1,
    const __bf16* __restrict__ W1, const __bf16* __restrict__ W2,
    const __bf16* __restrict__ W3,
    const float* __restrict__ b1v, const float* __restrict__ b2v,
    const float* __restrict__ gv,  const float* __restrict__ btv,
    const float* __restrict__ b3v,
    __bf16* __restrict__ outp, int mode)
{
    __shared__ alignas(32) __bf16 lds_h[16 * 512];   // bf16 A staging (16 KB)
    __shared__ alignas(16) float  lds_f[16 * 512];   // f32 scratch for LN (32 KB)
    __shared__ float lds_red[512];
    __shared__ float lds_mu[16];
    __shared__ float lds_rs[16];

    const int tid  = threadIdx.x;
    const int wave = tid >> 5;
    const int lane = tid & 31;
    const int m    = lane & 15;          // A-fragment row
    const int koff = (lane >> 4) * 16;   // K sub-offset: lane half h holds k = 16h..16h+15 (contiguous, matches packed B)

    const long rowBase = (long)blockIdx.x * 16;
    const long row = rowBase + m;
    long r0, r1;
    if (mode == 0) {
        long b = row / 49; int rem = (int)(row % 49);
        r0 = b * 7 + rem / 7;            // source node (row index)
        r1 = b * 7 + rem % 7;            // target node (col index)
    } else {
        r0 = row; r1 = row;
    }
    const __bf16* a0p = src0 + r0 * 512;
    const __bf16* a1p = src1 + r1 * 512;

    const v8f vzero = {};
    v8f acc[4];

    // ---------------- Stage 1: [16,1024] x [1024,512], +b1, ReLU ----------
#pragma unroll
    for (int t = 0; t < 4; ++t) acc[t] = vzero;
    {
        auto aPtr = [&](int kb) -> const __bf16* {
            return (kb < 16) ? (a0p + kb * 32) : (a1p + (kb - 16) * 32);
        };
        AFrag aCur = load_a_frag(aPtr(0), koff);
        AFrag aNxt;
        v16bf bCur[4], bNxt[4];
        load_b_frags(W1, 0, wave, lane, bCur);
        for (int kb = 0; kb < 32; kb += 2) {
            // issue k+1 loads before consuming k
            aNxt = load_a_frag(aPtr(kb + 1), koff);
            load_b_frags(W1, kb + 1, wave, lane, bNxt);
            if (kb + 3 < 32)  // keep weight stream warm in near caches
                __builtin_prefetch((const void*)(W1 + ((long)(kb + 3) * 32 + wave * 4) * 512 + lane * 16), 0, 3);
#pragma unroll
            for (int t = 0; t < 4; ++t) acc[t] = wmma_bf16(aCur, bCur[t], acc[t]);
            if (kb + 2 < 32) {
                aCur = load_a_frag(aPtr(kb + 2), koff);
                load_b_frags(W1, kb + 2, wave, lane, bCur);
            }
#pragma unroll
            for (int t = 0; t < 4; ++t) acc[t] = wmma_bf16(aNxt, bNxt[t], acc[t]);
        }
    }
#pragma unroll
    for (int t = 0; t < 4; ++t) {
        int col = wave * 64 + t * 16 + (lane & 15);
        float bb = b1v[col];
#pragma unroll
        for (int r = 0; r < 8; ++r) {
            int mm = r + (lane >> 4) * 8;
            float v = acc[t][r] + bb;
            lds_h[mm * 512 + col] = (__bf16)fmaxf(v, 0.f);
        }
    }
    __syncthreads();

    // ---------------- Stage 2: [16,512] x [512,512], +b2, LayerNorm, ReLU -
#pragma unroll
    for (int t = 0; t < 4; ++t) acc[t] = vzero;
    {
        const __bf16* aRow = &lds_h[m * 512];
        AFrag aCur = load_a_frag(aRow, koff);
        AFrag aNxt;
        v16bf bCur[4], bNxt[4];
        load_b_frags(W2, 0, wave, lane, bCur);
        for (int kb = 0; kb < 16; kb += 2) {
            aNxt = load_a_frag(aRow + (kb + 1) * 32, koff);
            load_b_frags(W2, kb + 1, wave, lane, bNxt);
#pragma unroll
            for (int t = 0; t < 4; ++t) acc[t] = wmma_bf16(aCur, bCur[t], acc[t]);
            if (kb + 2 < 16) {
                aCur = load_a_frag(aRow + (kb + 2) * 32, koff);
                load_b_frags(W2, kb + 2, wave, lane, bCur);
            }
#pragma unroll
            for (int t = 0; t < 4; ++t) acc[t] = wmma_bf16(aNxt, bNxt[t], acc[t]);
        }
    }
#pragma unroll
    for (int t = 0; t < 4; ++t) {
        int col = wave * 64 + t * 16 + (lane & 15);
        float bb = b2v[col];
#pragma unroll
        for (int r = 0; r < 8; ++r) {
            int mm = r + (lane >> 4) * 8;
            float v = acc[t][r] + bb;
            acc[t][r] = v;
            lds_f[mm * 512 + col] = v;
        }
    }
    __syncthreads();
    {   // two-level LN reduction: 16 threads per row, 32 cols each
        const float* rp = &lds_f[(tid >> 4) * 512 + (tid & 15) * 32];
        float s = 0.f, s2 = 0.f;
#pragma unroll
        for (int c = 0; c < 32; ++c) { float v = rp[c]; s += v; s2 += v * v; }
        lds_red[tid]       = s;
        lds_red[256 + tid] = s2;
    }
    __syncthreads();
    if (tid < 16) {
        float S = 0.f, S2 = 0.f;
#pragma unroll
        for (int k = 0; k < 16; ++k) {
            S  += lds_red[tid * 16 + k];
            S2 += lds_red[256 + tid * 16 + k];
        }
        float mu  = S * (1.f / 512.f);
        float var = S2 * (1.f / 512.f) - mu * mu;
        lds_mu[tid] = mu;
        lds_rs[tid] = rsqrtf(var + 1e-5f);
    }
    __syncthreads();
#pragma unroll
    for (int t = 0; t < 4; ++t) {
        int col = wave * 64 + t * 16 + (lane & 15);
        float gg = gv[col], bb = btv[col];
#pragma unroll
        for (int r = 0; r < 8; ++r) {
            int mm = r + (lane >> 4) * 8;
            float v = (acc[t][r] - lds_mu[mm]) * lds_rs[mm] * gg + bb;
            lds_h[mm * 512 + col] = (__bf16)fmaxf(v, 0.f);
        }
    }
    __syncthreads();

    // ---------------- Stage 3: [16,512] x [512,512], +b3 -> out (bf16) ----
#pragma unroll
    for (int t = 0; t < 4; ++t) acc[t] = vzero;
    {
        const __bf16* aRow = &lds_h[m * 512];
        AFrag aCur = load_a_frag(aRow, koff);
        AFrag aNxt;
        v16bf bCur[4], bNxt[4];
        load_b_frags(W3, 0, wave, lane, bCur);
        for (int kb = 0; kb < 16; kb += 2) {
            aNxt = load_a_frag(aRow + (kb + 1) * 32, koff);
            load_b_frags(W3, kb + 1, wave, lane, bNxt);
#pragma unroll
            for (int t = 0; t < 4; ++t) acc[t] = wmma_bf16(aCur, bCur[t], acc[t]);
            if (kb + 2 < 16) {
                aCur = load_a_frag(aRow + (kb + 2) * 32, koff);
                load_b_frags(W3, kb + 2, wave, lane, bCur);
            }
#pragma unroll
            for (int t = 0; t < 4; ++t) acc[t] = wmma_bf16(aNxt, bNxt[t], acc[t]);
        }
    }
#pragma unroll
    for (int t = 0; t < 4; ++t) {
        int col = wave * 64 + t * 16 + (lane & 15);
        float bb = b3v[col];
#pragma unroll
        for (int r = 0; r < 8; ++r) {
            int mm = r + (lane >> 4) * 8;
            float v = acc[t][r] + bb;
            outp[(rowBase + mm) * 512 + col] = (__bf16)v;
        }
    }
}

// ---------------------------------------------------------------------------
// Output heads: one wave per output row (node rows then edge rows).
// out layout: [B, n + n*n] with node_out first, edge_out after.
// ---------------------------------------------------------------------------
__global__ __launch_bounds__(256) void heads_kernel(
    const __bf16* __restrict__ node2, const __bf16* __restrict__ edge2,
    const float* __restrict__ fnw, const float* __restrict__ fnb,
    const float* __restrict__ few, const float* __restrict__ feb,
    float* __restrict__ out, long numNodes, long numEdges)
{
    const int wave = threadIdx.x >> 5;
    const int lane = threadIdx.x & 31;
    long r = (long)blockIdx.x * 8 + wave;
    if (r >= numNodes + numEdges) return;

    const __bf16* src;
    const float*  w;
    float bias;
    long  outIdx;
    const int n = 7, per = 7 + 49;
    if (r < numNodes) {
        src = node2 + r * 512; w = fnw; bias = fnb[0];
        long b = r / n; int i = (int)(r % n);
        outIdx = b * per + i;
    } else {
        long e = r - numNodes;
        src = edge2 + e * 512; w = few; bias = feb[0];
        long b = e / 49; int rem = (int)(e % 49);
        outIdx = b * per + n + rem;
    }
    float s = 0.f;
    for (int c = lane; c < 512; c += 32) s += (float)src[c] * w[c];
#pragma unroll
    for (int off = 16; off > 0; off >>= 1) s += __shfl_down(s, off, 32);
    if (lane == 0) out[outIdx] = s + bias;
}

// ---------------------------------------------------------------------------
extern "C" void kernel_launch(void* const* d_in, const int* in_sizes, int n_in,
                              void* d_out, int out_size, void* d_ws, size_t ws_size,
                              hipStream_t stream) {
    (void)in_sizes; (void)n_in; (void)out_size; (void)ws_size;

    const float* states = (const float*)d_in[0];
    const float* ew1 = (const float*)d_in[1];
    const float* eb1 = (const float*)d_in[2];
    const float* ew2 = (const float*)d_in[3];
    const float* eb2 = (const float*)d_in[4];
    const float* eg  = (const float*)d_in[5];
    const float* ebt = (const float*)d_in[6];
    const float* ew3 = (const float*)d_in[7];
    const float* eb3 = (const float*)d_in[8];
    const float* nw1 = (const float*)d_in[9];
    const float* nb1 = (const float*)d_in[10];
    const float* nw2 = (const float*)d_in[11];
    const float* nb2 = (const float*)d_in[12];
    const float* ng  = (const float*)d_in[13];
    const float* nbt = (const float*)d_in[14];
    const float* nw3 = (const float*)d_in[15];
    const float* nb3 = (const float*)d_in[16];
    const float* fnw = (const float*)d_in[17];
    const float* fnb = (const float*)d_in[18];
    const float* few = (const float*)d_in[19];
    const float* feb = (const float*)d_in[20];
    float* out = (float*)d_out;

    const long NODES = 4096L * 7;    // 28672
    const long EDGES = 4096L * 49;   // 200704
    const int  Hd = 512;

    char* ws = (char*)d_ws;
    size_t off = 0;
    auto alloc = [&](size_t bytes) -> char* {
        char* p = ws + off;
        off += (bytes + 255) & ~(size_t)255;
        return p;
    };
    __bf16* nodeB  = (__bf16*)alloc((size_t)NODES * Hd * 2);
    __bf16* node2B = (__bf16*)alloc((size_t)NODES * Hd * 2);
    __bf16* aggB   = (__bf16*)alloc((size_t)NODES * Hd * 2);
    __bf16* edgeB  = (__bf16*)alloc((size_t)EDGES * Hd * 2);  // reused for both edge passes
    __bf16* pew1 = (__bf16*)alloc(1024 * 512 * 2);
    __bf16* pew2 = (__bf16*)alloc(512 * 512 * 2);
    __bf16* pew3 = (__bf16*)alloc(512 * 512 * 2);
    __bf16* pnw1 = (__bf16*)alloc(1024 * 512 * 2);
    __bf16* pnw2 = (__bf16*)alloc(512 * 512 * 2);
    __bf16* pnw3 = (__bf16*)alloc(512 * 512 * 2);

    // 1) states -> bf16 node features
    {
        long ne = NODES * Hd;
        cvt_bf16_kernel<<<(unsigned)((ne + 255) / 256), 256, 0, stream>>>(states, nodeB, ne);
    }
    // 2) repack weights into WMMA-fragment-major bf16
    {
        long t1 = 1024L * 512, t2 = 512L * 512;
        pack_w_kernel<<<(unsigned)((t1 + 255) / 256), 256, 0, stream>>>(ew1, pew1, 1024, 512);
        pack_w_kernel<<<(unsigned)((t2 + 255) / 256), 256, 0, stream>>>(ew2, pew2, 512, 512);
        pack_w_kernel<<<(unsigned)((t2 + 255) / 256), 256, 0, stream>>>(ew3, pew3, 512, 512);
        pack_w_kernel<<<(unsigned)((t1 + 255) / 256), 256, 0, stream>>>(nw1, pnw1, 1024, 512);
        pack_w_kernel<<<(unsigned)((t2 + 255) / 256), 256, 0, stream>>>(nw2, pnw2, 512, 512);
        pack_w_kernel<<<(unsigned)((t2 + 255) / 256), 256, 0, stream>>>(nw3, pnw3, 512, 512);
    }
    // 3) edge MLP pass 1  (A = concat(node[row], node[col]))
    fused_mlp3<<<(unsigned)(EDGES / 16), 256, 0, stream>>>(
        nodeB, nodeB, pew1, pew2, pew3, eb1, eb2, eg, ebt, eb3, edgeB, /*mode=*/0);
    // 4) segment-sum over the 7 contiguous edges per source node
    {
        long ne = NODES * Hd;
        agg_kernel<<<(unsigned)((ne + 255) / 256), 256, 0, stream>>>(edgeB, aggB, ne);
    }
    // 5) node MLP  (A = concat(node, agg))
    fused_mlp3<<<(unsigned)(NODES / 16), 256, 0, stream>>>(
        nodeB, aggB, pnw1, pnw2, pnw3, nb1, nb2, ng, nbt, nb3, node2B, /*mode=*/1);
    // 6) edge MLP pass 2 on node2 (overwrites edgeB)
    fused_mlp3<<<(unsigned)(EDGES / 16), 256, 0, stream>>>(
        node2B, node2B, pew1, pew2, pew3, eb1, eb2, eg, ebt, eb3, edgeB, /*mode=*/0);
    // 7) output heads
    {
        long rows = NODES + EDGES;                 // 229376, divisible by 8
        heads_kernel<<<(unsigned)(rows / 8), 256, 0, stream>>>(
            node2B, edgeB, fnw, fnb, few, feb, out, NODES, EDGES);
    }
}
